// DeepseekV4MLAAttention_26577257628069
// MI455X (gfx1250) — matrix operand order repaired and checked by reading, not compiled, not run on hardware
//
#include <hip/hip_runtime.h>
#include <hip/hip_bf16.h>
#include <math.h>

// ---------------- problem constants (from reference) ----------------
#define NSEQ   64
#define DIM    4096
#define NH     32
#define HD     192
#define RD     64
#define QLR    1536
#define OLR    512
#define NG     8
#define BS     64
#define MAXB   64
#define LMAX   4096
#define TOPK_K 1024
#define EPSV   1e-6f
#define SCALEV 0.07216878364870323f   // 192^-0.5
#define NEGV   -1e30f

typedef __attribute__((ext_vector_type(16))) __bf16 v16bf;
typedef __attribute__((ext_vector_type(8)))  float  v8f;

// order-preserving key: larger float -> larger unsigned
__device__ __forceinline__ unsigned okey(float f) {
  unsigned u = __float_as_uint(f);
  return (u & 0x80000000u) ? ~u : (u | 0x80000000u);
}
__device__ __forceinline__ float okey_inv(unsigned k) {
  return (k & 0x80000000u) ? __uint_as_float(k & 0x7fffffffu)
                           : __uint_as_float(~k);
}

// ============================================================================
// Generic C = A * B^T with bf16 WMMA (one wave per 16x16 C tile).
// A: (M x K) row-major, lda.  B: (N x K) row-major, ldb (i.e. weight (out,in)).
// z-strides allow grouped GEMMs (wo_a).
// Lane layouts used here (A/B in contiguous k order, not the ISA table order):
//   A 16x32 bf16 : lane L holds M=L%16; vgpr j   -> K=16*(L/16)+2j
//   B 32x16 bf16 : lane L holds N=L%16; vgpr j   -> K=16*(L/16)+2j
//   D 16x16 f32  : lane L col N=L%16; vgpr r -> row M = r + 8*(L/16)
// ============================================================================
__global__ void gemm_abt_bf16(const float* __restrict__ A, int lda, long sAz,
                              const float* __restrict__ B, int ldb, long sBz,
                              float* __restrict__ C, int ldc, long sCz, int K)
{
  const int lane = threadIdx.x;
  const int r = lane & 15, half = lane >> 4;
  const float* Ap = A + (long)blockIdx.z * sAz + (long)(blockIdx.y * 16 + r) * lda;
  const float* Bp = B + (long)blockIdx.z * sBz + (long)(blockIdx.x * 16 + r) * ldb;
  v8f acc = {};
  for (int k0 = 0; k0 < K; k0 += 32) {
    if (k0 + 128 < K) {
      __builtin_prefetch(Ap + k0 + 128, 0, 1);
      __builtin_prefetch(Bp + k0 + 128, 0, 1);
    }
    v16bf a, b;
#pragma unroll
    for (int j = 0; j < 4; ++j) {
      int ka = k0 + 16 * half + 2 * j;
      a[2 * j]     = (__bf16)Ap[ka];
      a[2 * j + 1] = (__bf16)Ap[ka + 1];
      a[2 * j + 8] = (__bf16)Ap[ka + 8];
      a[2 * j + 9] = (__bf16)Ap[ka + 9];
    }
#pragma unroll
    for (int j = 0; j < 8; ++j) {
      int kb = k0 + 16 * half + 2 * j;
      b[2 * j]     = (__bf16)Bp[kb];
      b[2 * j + 1] = (__bf16)Bp[kb + 1];
    }
    acc = __builtin_amdgcn_wmma_f32_16x16x32_bf16(false, a, false, b,
                                                  (short)0, acc, false, false);
  }
  float* Cp = C + (long)blockIdx.z * sCz + (long)blockIdx.x * 16;
#pragma unroll
  for (int vr = 0; vr < 8; ++vr) {
    int m = blockIdx.y * 16 + vr + 8 * half;
    Cp[(long)m * ldc + r] = acc[vr];
  }
}

// ============================================================================
// kv = rope(rms(x @ wkv^T, kv_norm)) -> kvn  (one block per sequence)
// ============================================================================
__global__ void kv_proj(const float* __restrict__ x, const float* __restrict__ wkv,
                        const float* __restrict__ kvw, const int* __restrict__ ctxs,
                        float* __restrict__ kvn)
{
  __shared__ float v[HD];
  __shared__ float red[256];
  const int n = blockIdx.x, t = threadIdx.x;
  const float* xr = x + (long)n * DIM;
  if (t < HD) {
    const float* wr = wkv + (long)t * DIM;
    float s = 0.f;
    for (int k = 0; k < DIM; k += 4)
      s += xr[k] * wr[k] + xr[k + 1] * wr[k + 1] +
           xr[k + 2] * wr[k + 2] + xr[k + 3] * wr[k + 3];
    v[t] = s;
  }
  __syncthreads();
  red[t] = (t < HD) ? v[t] * v[t] : 0.f;
  __syncthreads();
  for (int st = 128; st; st >>= 1) { if (t < st) red[t] += red[t + st]; __syncthreads(); }
  const float inv = rsqrtf(red[0] / (float)HD + EPSV);
  if (t < HD) v[t] = v[t] * inv * kvw[t];
  __syncthreads();
  if (t < RD / 2) {
    const float pos = (float)(ctxs[n] - 1);
    const float invf = powf(10000.f, -(2.f * (float)t) / (float)RD);
    float sn, cs; sincosf(pos * invf, &sn, &cs);
    const int i0 = HD - RD + 2 * t;
    float x0 = v[i0], x1 = v[i0 + 1];
    v[i0]     = x0 * cs - x1 * sn;
    v[i0 + 1] = x0 * sn + x1 * cs;
  }
  __syncthreads();
  if (t < HD) kvn[(long)n * HD + t] = v[t];
}

// rms-normalize rows in place, scale by w   (one block per row)
__global__ void rms_rows(float* __restrict__ buf, const float* __restrict__ w, int cols)
{
  __shared__ float red[256];
  const int n = blockIdx.x, t = threadIdx.x;
  float* row = buf + (long)n * cols;
  float ss = 0.f;
  for (int i = t; i < cols; i += 256) { float v = row[i]; ss += v * v; }
  red[t] = ss; __syncthreads();
  for (int st = 128; st; st >>= 1) { if (t < st) red[t] += red[t + st]; __syncthreads(); }
  const float inv = rsqrtf(red[0] / (float)cols + EPSV);
  for (int i = t; i < cols; i += 256) row[i] = row[i] * inv * w[i];
}

// rope last RD dims of each q head  (grid: (NH, NSEQ), 32 threads)
__global__ void rope_q(float* __restrict__ q, const int* __restrict__ ctxs)
{
  const int h = blockIdx.x, n = blockIdx.y, t = threadIdx.x;
  float* row = q + ((long)n * NH + h) * HD + (HD - RD);
  const float pos = (float)(ctxs[n] - 1);
  const float invf = powf(10000.f, -(2.f * (float)t) / (float)RD);
  float sn, cs; sincosf(pos * invf, &sn, &cs);
  float x0 = row[2 * t], x1 = row[2 * t + 1];
  row[2 * t]     = x0 * cs - x1 * sn;
  row[2 * t + 1] = x0 * sn + x1 * cs;
}

// ============================================================================
// logits[n,h,l] = SCALE * q[n,h,:] . kv[l,:]  (masked to NEG for l >= ctx)
// grid (LMAX/16, NH/16, NSEQ), one wave per block.
// ============================================================================
__global__ void logits_wmma(const float* __restrict__ q, const float* __restrict__ cache,
                            const float* __restrict__ kvn, const int* __restrict__ bt,
                            const int* __restrict__ ctxs, float* __restrict__ logits)
{
  const int n = blockIdx.z, htile = blockIdx.y, ltile = blockIdx.x;
  const int lane = threadIdx.x, r = lane & 15, half = lane >> 4;
  const int ctx = ctxs[n];
  const int l = ltile * 16 + r;
  const bool valid = (l < ctx);
  const float* kvrow = cache;  // safe default
  if (valid) {
    if (l == ctx - 1) kvrow = kvn + (long)n * HD;   // freshly-written slot
    else {
      int slot = bt[n * MAXB + (l >> 6)] * BS + (l & 63);
      kvrow = cache + (long)slot * HD;
    }
  }
  const float* qrow = q + ((long)n * NH + htile * 16 + r) * HD;
  v8f acc = {};
#pragma unroll
  for (int k0 = 0; k0 < HD; k0 += 32) {
    v16bf a, b;
#pragma unroll
    for (int j = 0; j < 4; ++j) {
      int ka = k0 + 16 * half + 2 * j;
      a[2 * j]     = (__bf16)qrow[ka];
      a[2 * j + 1] = (__bf16)qrow[ka + 1];
      a[2 * j + 8] = (__bf16)qrow[ka + 8];
      a[2 * j + 9] = (__bf16)qrow[ka + 9];
    }
#pragma unroll
    for (int j = 0; j < 8; ++j) {
      int kb = k0 + 16 * half + 2 * j;
      b[2 * j]     = (__bf16)(valid ? kvrow[kb]     : 0.f);
      b[2 * j + 1] = (__bf16)(valid ? kvrow[kb + 1] : 0.f);
    }
    acc = __builtin_amdgcn_wmma_f32_16x16x32_bf16(false, a, false, b,
                                                  (short)0, acc, false, false);
  }
#pragma unroll
  for (int vr = 0; vr < 8; ++vr) {
    int hh = htile * 16 + vr + 8 * half;
    logits[((long)n * NH + hh) * LMAX + l] = valid ? acc[vr] * SCALEV : NEGV;
  }
}

// ============================================================================
// Exact top-k(1024) + sink softmax per (n,h) row, probs written in place.
// ============================================================================
__global__ void __launch_bounds__(256)
topk_softmax(float* __restrict__ logits, const float* __restrict__ sinkv)
{
  __shared__ float    row[LMAX];
  __shared__ unsigned hist[256];
  __shared__ float    red[256];
  __shared__ int      tiecnt[256];
  __shared__ int      tiebase[256];
  __shared__ unsigned s_bucket;
  __shared__ int      s_want;

  const int h = blockIdx.x, n = blockIdx.y, t = threadIdx.x;
  float* grow = logits + ((long)n * NH + h) * LMAX;

  float lmax = -3.0e38f;
  for (int i = t; i < LMAX; i += 256) { float v = grow[i]; row[i] = v; lmax = fmaxf(lmax, v); }
  red[t] = lmax; __syncthreads();
  for (int st = 128; st; st >>= 1) { if (t < st) red[t] = fmaxf(red[t], red[t + st]); __syncthreads(); }
  const float vmax = red[0];
  __syncthreads();

  unsigned prefix = 0;
  int want = TOPK_K;
  for (int pass = 0; pass < 4; ++pass) {
    const int shift = 24 - 8 * pass;
    hist[t] = 0u; __syncthreads();
    for (int i = t; i < LMAX; i += 256) {
      unsigned key = okey(row[i]);
      if (pass == 0 || (key >> (shift + 8)) == prefix)
        atomicAdd(&hist[(key >> shift) & 255u], 1u);
    }
    __syncthreads();
    if (t == 0) {
      int acc = 0, b = 255;
      for (; b >= 0; --b) { acc += (int)hist[b]; if (acc >= want) break; }
      if (b < 0) b = 0;
      s_bucket = (unsigned)b;
      s_want   = want - (acc - (int)hist[b]);  // #elements needed inside bucket b
    }
    __syncthreads();
    prefix = (prefix << 8) | s_bucket;
    want   = s_want;
    __syncthreads();
  }
  const unsigned tkey = prefix;        // key of the k-th largest value
  const float vT   = okey_inv(tkey);
  const float sink = sinkv[h];
  const float m    = fmaxf(vmax, sink);

  float lsum = 0.f;
  for (int i = t; i < LMAX; i += 256) {
    unsigned key = okey(row[i]);
    if (key > tkey) lsum += __expf(row[i] - m);
  }
  red[t] = lsum; __syncthreads();
  for (int st = 128; st; st >>= 1) { if (t < st) red[t] += red[t + st]; __syncthreads(); }
  const float eT    = __expf(vT - m);
  const float denom = red[0] + (float)want * eT + __expf(sink - m);
  const float pT    = eT / denom;

  // index-ordered tie ranking over contiguous 16-element chunks
  const int i0 = t * (LMAX / 256);
  int cnt = 0;
  for (int i = i0; i < i0 + (LMAX / 256); ++i) if (okey(row[i]) == tkey) ++cnt;
  tiecnt[t] = cnt; __syncthreads();
  if (t == 0) { int a = 0; for (int i = 0; i < 256; ++i) { tiebase[i] = a; a += tiecnt[i]; } }
  __syncthreads();
  int rank = tiebase[t];
  for (int i = i0; i < i0 + (LMAX / 256); ++i) {
    unsigned key = okey(row[i]);
    float p = 0.f;
    if (key > tkey) p = __expf(row[i] - m) / denom;
    else if (key == tkey) { if (rank < want) p = pT; ++rank; }
    grow[i] = p;
  }
}

// ============================================================================
// o[n,h,d] = sum_l probs[n,h,l] * kv[l,d]
// grid (HD/16, NH/16, NSEQ), one wave per block.
// kv tile (32 pos x 16 dims) staged into LDS with the CDNA5 async copy path:
// each lane owns one kv row (64 B) and issues 4x GLOBAL_LOAD_ASYNC_TO_LDS_B128
// (inst_offset is added to BOTH the global and LDS addresses, ISA 10.x async
// pseudocode), then a single s_wait_asynccnt 0 publishes the tile to the wave.
// s_wait_dscnt 0 before re-issue guarantees the previous tile's ds_loads have
// landed in VGPRs before the async engine overwrites LDS.
// ============================================================================
__global__ void attn_out_wmma(const float* __restrict__ probs, const float* __restrict__ cache,
                              const float* __restrict__ kvn, const int* __restrict__ bt,
                              const int* __restrict__ ctxs, float* __restrict__ o)
{
  __shared__ float kvt[32][16];   // 2 KB tile, 64 B contiguous per lane-row
  const int n = blockIdx.z, htile = blockIdx.y, dtile = blockIdx.x;
  const int lane = threadIdx.x, r = lane & 15, half = lane >> 4;
  const int ctx = ctxs[n];
  const float* prow = probs + ((long)n * NH + htile * 16 + r) * LMAX;
  const unsigned lds_row = (unsigned)(size_t)(&kvt[lane][0]);  // AS(3) byte offset
  v8f acc = {};
  for (int k0 = 0; k0 < LMAX; k0 += 32) {
    const int pos = k0 + lane;
    const float* kvrow;
    if (pos == ctx - 1) kvrow = kvn + (long)n * HD;
    else kvrow = cache + (long)(bt[n * MAXB + (pos >> 6)] * BS + (pos & 63)) * HD;
    const float* src = kvrow + dtile * 16;
    // previous iteration's LDS reads must be in VGPRs before overwrite
    asm volatile("s_wait_dscnt 0" ::: "memory");
    asm volatile("global_load_async_to_lds_b128 %0, %1, off"
                 :: "v"(lds_row), "v"(src) : "memory");
    asm volatile("global_load_async_to_lds_b128 %0, %1, off offset:16"
                 :: "v"(lds_row), "v"(src) : "memory");
    asm volatile("global_load_async_to_lds_b128 %0, %1, off offset:32"
                 :: "v"(lds_row), "v"(src) : "memory");
    asm volatile("global_load_async_to_lds_b128 %0, %1, off offset:48"
                 :: "v"(lds_row), "v"(src) : "memory");
    asm volatile("s_wait_asynccnt 0" ::: "memory");
    v16bf a, b;
#pragma unroll
    for (int j = 0; j < 4; ++j) {
      int ka = k0 + 16 * half + 2 * j;
      a[2 * j]     = (__bf16)prow[ka];
      a[2 * j + 1] = (__bf16)prow[ka + 1];
      a[2 * j + 8] = (__bf16)prow[ka + 8];
      a[2 * j + 9] = (__bf16)prow[ka + 9];
    }
#pragma unroll
    for (int j = 0; j < 8; ++j) {
      int kb = 16 * half + 2 * j;
      b[2 * j]     = (__bf16)kvt[kb][r];
      b[2 * j + 1] = (__bf16)kvt[kb + 1][r];
    }
    acc = __builtin_amdgcn_wmma_f32_16x16x32_bf16(false, a, false, b,
                                                  (short)0, acc, false, false);
  }
#pragma unroll
  for (int vr = 0; vr < 8; ++vr) {
    int hh = htile * 16 + vr + 8 * half;
    o[((long)n * NH + hh) * HD + dtile * 16 + r] = acc[vr];
  }
}

// ============================================================================
extern "C" void kernel_launch(void* const* d_in, const int* in_sizes, int n_in,
                              void* d_out, int out_size, void* d_ws, size_t ws_size,
                              hipStream_t stream)
{
  const float* x       = (const float*)d_in[0];
  const float* cache   = (const float*)d_in[1];
  const int*   bt      = (const int*)d_in[2];
  const int*   ctx     = (const int*)d_in[3];
  // d_in[4] slot_mapping: implied by block_tables + (ctx-1); unused
  const float* wq_a    = (const float*)d_in[5];
  const float* q_norm  = (const float*)d_in[6];
  const float* wq_b    = (const float*)d_in[7];
  const float* wkv     = (const float*)d_in[8];
  const float* kv_norm = (const float*)d_in[9];
  const float* wo_a    = (const float*)d_in[10];
  const float* wo_b    = (const float*)d_in[11];
  const float* sink    = (const float*)d_in[12];
  float* out = (float*)d_out;

  float* ws     = (float*)d_ws;
  float* qa     = ws;                               // NSEQ*QLR
  float* qbuf   = qa     + (size_t)NSEQ * QLR;      // NSEQ*NH*HD
  float* kvn    = qbuf   + (size_t)NSEQ * NH * HD;  // NSEQ*HD
  float* obuf   = kvn    + (size_t)NSEQ * HD;       // NSEQ*NH*HD
  float* lat    = obuf   + (size_t)NSEQ * NH * HD;  // NSEQ*NG*OLR
  float* logits = lat    + (size_t)NSEQ * NG * OLR; // NSEQ*NH*LMAX  (~34 MB, L2-resident)

  // 1. kv path (rms + rope), cached-row substitute
  kv_proj<<<NSEQ, 256, 0, stream>>>(x, wkv, kv_norm, ctx, kvn);
  // 2. qa = x @ wq_a^T ; rms ; q = qa @ wq_b^T ; rope
  gemm_abt_bf16<<<dim3(QLR / 16, NSEQ / 16, 1), 32, 0, stream>>>(
      x, DIM, 0, wq_a, DIM, 0, qa, QLR, 0, DIM);
  rms_rows<<<NSEQ, 256, 0, stream>>>(qa, q_norm, QLR);
  gemm_abt_bf16<<<dim3(NH * HD / 16, NSEQ / 16, 1), 32, 0, stream>>>(
      qa, QLR, 0, wq_b, QLR, 0, qbuf, NH * HD, 0, QLR);
  rope_q<<<dim3(NH, NSEQ), RD / 2, 0, stream>>>(qbuf, ctx);
  // 3. attention: logits -> top-k softmax (in place) -> output
  logits_wmma<<<dim3(LMAX / 16, NH / 16, NSEQ), 32, 0, stream>>>(
      qbuf, cache, kvn, bt, ctx, logits);
  topk_softmax<<<dim3(NH, NSEQ), 256, 0, stream>>>(logits, sink);
  attn_out_wmma<<<dim3(HD / 16, NH / 16, NSEQ), 32, 0, stream>>>(
      logits, cache, kvn, bt, ctx, obuf);
  // 4. grouped wo_a then wo_b
  gemm_abt_bf16<<<dim3(OLR / 16, NSEQ / 16, NG), 32, 0, stream>>>(
      obuf, NH * HD, 768, wo_a, 768, (long)OLR * 768, lat, NG * OLR, OLR, 768);
  gemm_abt_bf16<<<dim3(NG * OLR / 16, NSEQ / 16, 1), 32, 0, stream>>>(
      lat, NG * OLR, 0, wo_b, NG * OLR, 0, out, DIM, 0, NG * OLR);
}
